// ProteinConditionedEGNNDynamics_53644141527275
// MI455X (gfx1250) — matrix operand order repaired and checked by reading, compiled, not run on hardware
//
#include <hip/hip_runtime.h>
#include <hip/hip_bf16.h>

#define BS   4
#define NL   128
#define NPR  1024
#define LNF_ 128
#define PNF_ 128
#define HD   128
#define THRESH 10.0f
#define NORMC  100.0f
#define EPSC   1e-8f

typedef __attribute__((ext_vector_type(16))) _Float16 v16h;
typedef __attribute__((ext_vector_type(8)))  float    v8f;

__device__ __forceinline__ float rcp_f(float x)  { return __builtin_amdgcn_rcpf(x); }
__device__ __forceinline__ float sigm_f(float x) { return rcp_f(1.f + __expf(-x)); }
__device__ __forceinline__ float silu_f(float x) { return x * rcp_f(1.f + __expf(-x)); }
__device__ __forceinline__ float tanh_f(float x) {
  float a = fabsf(x);
  float t = __expf(-2.f * a);                 // t in (0,1], no overflow
  float r = (1.f - t) * rcp_f(1.f + t);
  return copysignf(r, x);
}

// ---------------- phase 1: per-row linear projections ----------------
__global__ void egnn_row_precompute(const float* __restrict__ hlig,
                                    const float* __restrict__ hprot,
                                    const float* __restrict__ aW1, const float* __restrict__ ab1,
                                    const float* __restrict__ vW1, const float* __restrict__ vb1,
                                    const float* __restrict__ cW1, const float* __restrict__ cb1,
                                    float* __restrict__ HPa, float* __restrict__ HPv, float* __restrict__ HPc,
                                    float* __restrict__ HLa, float* __restrict__ HLc)
{
  __shared__ float row[HD];
  const int r = blockIdx.x;
  const int h = threadIdx.x;
  if (r < BS*NPR) {
    row[h] = hprot[(size_t)r*PNF_ + h];
    __syncthreads();
    float aA = ab1[h], aV = vb1[h], aC = cb1[h];
    #pragma unroll 8
    for (int k = 0; k < PNF_; ++k) {
      float x = row[k];
      aA += x * aW1[(LNF_ + k)*HD + h];
      aV += x * vW1[k*HD + h];
      aC += x * cW1[(LNF_ + k)*HD + h];
    }
    HPa[(size_t)r*HD + h] = aA;
    HPv[(size_t)r*HD + h] = aV;
    HPc[(size_t)r*HD + h] = aC;
  } else {
    const int q = r - BS*NPR;
    row[h] = hlig[(size_t)q*LNF_ + h];
    __syncthreads();
    float aA = 0.f, aC = 0.f;
    #pragma unroll 8
    for (int k = 0; k < LNF_; ++k) {
      float x = row[k];
      aA += x * aW1[k*HD + h];
      aC += x * cW1[k*HD + h];
    }
    HLa[(size_t)q*HD + h] = aA;
    HLc[(size_t)q*HD + h] = aC;
  }
}

// ---------------- phase 1b: swizzle aW2 / vW2 into WMMA B-fragment layout ----------------
// B 32x16 f16: lanes 0-15: N=lane, K=0..7,16..23; lanes 16-31: N=lane-16, K=8..15,24..31
__global__ void egnn_swizzle_b(const float* __restrict__ aW2,
                               const float* __restrict__ vW2,
                               _Float16* __restrict__ wsB)
{
  int q = blockIdx.x*blockDim.x + threadIdx.x;   // 0..2047
  if (q >= 2048) return;
  int layer = q >> 10, rem = q & 1023;
  int frag = rem >> 5, l = rem & 31;
  int kt = frag >> 3, nt = frag & 7;
  int kbB = (l & 16) ? 8 : 0;
  int n   = nt*16 + (l & 15);
  const float* W = layer ? vW2 : aW2;
  _Float16* dst = wsB + layer*16384 + frag*512 + l*16;
  #pragma unroll
  for (int idx = 0; idx < 16; ++idx)
    dst[idx] = (_Float16)W[(kt*32 + kbB + (idx & 7) + ((idx & 8) << 1))*HD + n];
}

__device__ __forceinline__ v16h load_bfrag(const _Float16* __restrict__ base,
                                           int frag, int lane)
{
  union { v16h v; float4 f[2]; } u;
  const float4* p = (const float4*)(base + frag*512 + lane*16);
  u.f[0] = p[0];
  u.f[1] = p[1];
  return u.v;
}

// ---------------- phase 2: fused per-pair MLP + reductions ----------------
__global__ void __launch_bounds__(256)
egnn_cross_main(const float* __restrict__ x_ligand, const float* __restrict__ x_protein,
                const float* __restrict__ ligand_mask, const float* __restrict__ protein_mask,
                const float* __restrict__ aW1, const float* __restrict__ ab2,
                const float* __restrict__ aW3, const float* __restrict__ ab3p,
                const float* __restrict__ vW1, const float* __restrict__ vb2,
                const float* __restrict__ cW1, const float* __restrict__ cW2,
                const float* __restrict__ cb2p,
                const float* __restrict__ HPa, const float* __restrict__ HPv, const float* __restrict__ HPc,
                const float* __restrict__ HLa, const float* __restrict__ HLc,
                const _Float16* __restrict__ wsB,
                float* __restrict__ out)
{
  __shared__ float sH[16*HD];          // 8 KB  cross-wave h accumulator
  __shared__ float sX[16*3];
  __shared__ float sHLa[16*HD];        // 8 KB  ligand-tile a-projection
  __shared__ float sHLc[16*HD];        // 8 KB  ligand-tile c-projection
  __shared__ float sWa[HD], sWv[HD], sWc[HD], sC2[HD];   // ds-rows + cW2

  const int JCH = 4, JCHUNK = NPR/JCH, ITILES = NL/16;
  int bi = blockIdx.x;
  int jc = bi % JCH;  int rest = bi / JCH;
  int it = rest % ITILES;  int b = rest / ITILES;
  int i0 = it*16, jbase = jc*JCHUNK;
  int tid = threadIdx.x;
  int wave = tid >> 5, lane = tid & 31;

  // stage block invariants into LDS
  for (int t = tid; t < 16*HD; t += 256) {
    sH[t] = 0.f;
    int m = t >> 7, h = t & 127;
    sHLa[t] = HLa[(size_t)(b*NL + i0 + m)*HD + h];
    sHLc[t] = HLc[(size_t)(b*NL + i0 + m)*HD + h];
  }
  if (tid < 48) sX[tid] = 0.f;
  if (tid < HD) {
    sWa[tid] = aW1[256*HD + tid];
    sWv[tid] = vW1[128*HD + tid];
    sWc[tid] = cW1[256*HD + tid];
    sC2[tid] = cW2[tid];
  }
  __syncthreads();

  const int lan  = lane & 15;
  const int kb   = (lane & 16) ? 8 : 0;   // A-fragment K base within each 32-K step
  const int offD = (lane & 16) ? 8 : 0;   // D-layout row offset
  const float ab3 = ab3p[0];
  const float cb2 = cb2p[0];
  const _Float16* Ba = wsB;
  const _Float16* Bv = wsB + 16384;

  const int irow = b*NL + i0 + lan;
  const float xlx = x_ligand[irow*3+0];
  const float xly = x_ligand[irow*3+1];
  const float xlz = x_ligand[irow*3+2];
  const float* hla = &sHLa[lan*HD];
  const float* hlc = &sHLc[lan*HD];
  float ab2v[8], aW3v[8], vb2v[8];
  #pragma unroll
  for (int nt = 0; nt < 8; ++nt) {
    int hh = nt*16 + lan;
    ab2v[nt] = ab2[hh]; aW3v[nt] = aW3[hh]; vb2v[nt] = vb2[hh];
  }

  float hacc[64];
  #pragma unroll
  for (int t = 0; t < 64; ++t) hacc[t] = 0.f;
  float xa0 = 0.f, xa1 = 0.f, xa2 = 0.f;

  for (int j = jbase + wave; j < jbase + JCHUNK; j += 8) {
    // keep the loop body's memory loads inside the loop (no LICM -> no spills)
    asm volatile("" ::: "memory");

    const int jrow = b*NPR + j;
    const float px = x_protein[jrow*3+0];
    const float py = x_protein[jrow*3+1];
    const float pz = x_protein[jrow*3+2];
    const float pm = protein_mask[jrow];
    const float rx = xlx - px, ry = xly - py, rz = xlz - pz;
    const float dsown = rx*rx + ry*ry + rz*rz;

    float edge8[8];
    #pragma unroll
    for (int r = 0; r < 8; ++r) {
      float dsr = __shfl(dsown, offD + r, 32);             // ds for i = r + offD
      edge8[r] = (dsr + EPSC < THRESH*THRESH) ? 1.f : 0.f; // dist<10  <=>  ds+eps<100
    }

    const float* hpa = HPa + (size_t)jrow*HD;
    const float* hpv = HPv + (size_t)jrow*HD;
    const float* hpc = HPc + (size_t)jrow*HD;

    // ---- build a1 in A-fragment layout; accumulate coord-gate dot ----
    v16h aF[4];
    float cacc = 0.f;
    #pragma unroll
    for (int kt = 0; kt < 4; ++kt) {
      v16h af;
      #pragma unroll
      for (int run = 0; run < 2; ++run) {
        int hh = kt*32 + kb + run*16;
        #pragma unroll
        for (int qq = 0; qq < 8; ++qq) {
          int h = hh + qq;
          float za = hla[h] + hpa[h] + dsown*sWa[h];
          float zc = hlc[h] + hpc[h] + dsown*sWc[h];
          af[run*8+qq] = (_Float16)silu_f(za);
          cacc += silu_f(zc) * sC2[h];
        }
      }
      aF[kt] = af;
    }

    // coord gate: lanes l and l+16 cover complementary h's for i = lan
    float cfull = cacc + __shfl_xor(cacc, 16, 32);
    float cwgt  = tanh_f(cfull + cb2) * pm;

    // ---- attention layer 2 (a1 @ aW2) + silu + attn-head dot ----
    float acc8[8];
    #pragma unroll
    for (int r = 0; r < 8; ++r) acc8[r] = 0.f;
    #pragma unroll
    for (int nt = 0; nt < 8; ++nt) {
      v8f d = {};
      #pragma unroll
      for (int kt = 0; kt < 4; ++kt) {
        v16h bf = load_bfrag(Ba, kt*8 + nt, lane);
        d = __builtin_amdgcn_wmma_f32_16x16x32_f16(false, aF[kt], false, bf,
                                                   (short)0, d, false, false);
      }
      #pragma unroll
      for (int r = 0; r < 8; ++r) {
        float z = d[r] + ab2v[nt];
        acc8[r] += silu_f(z) * aW3v[nt];
      }
    }
    #pragma unroll
    for (int m = 1; m < 16; m <<= 1) {
      #pragma unroll
      for (int r = 0; r < 8; ++r) acc8[r] += __shfl_xor(acc8[r], m, 32);
    }
    float attnE[8];
    #pragma unroll
    for (int r = 0; r < 8; ++r)
      attnE[r] = sigm_f(acc8[r] + ab3) * pm * edge8[r];

    // ---- build v1 fragments (aF dead now) ----
    v16h vF[4];
    #pragma unroll
    for (int kt = 0; kt < 4; ++kt) {
      v16h vf;
      #pragma unroll
      for (int run = 0; run < 2; ++run) {
        int hh = kt*32 + kb + run*16;
        #pragma unroll
        for (int qq = 0; qq < 8; ++qq) {
          int h = hh + qq;
          vf[run*8+qq] = (_Float16)silu_f(hpv[h] + dsown*sWv[h]);
        }
      }
      vF[kt] = vf;
    }

    // ---- value projection (v1 @ vW2) + gated accumulation ----
    #pragma unroll
    for (int nt = 0; nt < 8; ++nt) {
      v8f d = {};
      #pragma unroll
      for (int kt = 0; kt < 4; ++kt) {
        v16h bf = load_bfrag(Bv, kt*8 + nt, lane);
        d = __builtin_amdgcn_wmma_f32_16x16x32_f16(false, vF[kt], false, bf,
                                                   (short)0, d, false, false);
      }
      #pragma unroll
      for (int r = 0; r < 8; ++r)
        hacc[nt*8 + r] += attnE[r] * (d[r] + vb2v[nt]);
    }

    // ---- coordinates: lanes 0..15 own ligand i = lane ----
    if (lane < 16) {
      if (dsown + EPSC < THRESH*THRESH) {
        float dist = sqrtf(dsown + EPSC);
        float g = cwgt * rcp_f(dist + EPSC);
        xa0 += rx*g; xa1 += ry*g; xa2 += rz*g;
      }
    }
  }

  // combine waves via LDS
  #pragma unroll
  for (int nt = 0; nt < 8; ++nt) {
    #pragma unroll
    for (int r = 0; r < 8; ++r) {
      int m = r + offD, h = nt*16 + lan;
      atomicAdd(&sH[m*HD + h], hacc[nt*8 + r]);
    }
  }
  if (lane < 16) {
    atomicAdd(&sX[lane*3+0], xa0);
    atomicAdd(&sX[lane*3+1], xa1);
    atomicAdd(&sX[lane*3+2], xa2);
  }
  __syncthreads();

  float* out_h = out;
  float* out_x = out + (size_t)BS*NL*HD;
  for (int t = tid; t < 16*HD; t += 256) {
    int m = t >> 7, h = t & 127;
    int gi = b*NL + i0 + m;
    float lm = ligand_mask[gi];
    atomicAdd(&out_h[(size_t)gi*HD + h], sH[t] * (lm * (1.f/NORMC)));
  }
  if (tid < 48) {
    int m = tid/3, c = tid%3;
    int gi = b*NL + i0 + m;
    float lm = ligand_mask[gi];
    atomicAdd(&out_x[(size_t)gi*3 + c], sX[tid] * (lm * (1.f/NORMC)));
  }
}

extern "C" void kernel_launch(void* const* d_in, const int* in_sizes, int n_in,
                              void* d_out, int out_size, void* d_ws, size_t ws_size,
                              hipStream_t stream)
{
  const float* h_ligand    = (const float*)d_in[0];
  const float* x_ligand    = (const float*)d_in[1];
  const float* h_protein   = (const float*)d_in[2];
  const float* x_protein   = (const float*)d_in[3];
  const float* ligand_mask = (const float*)d_in[4];
  const float* protein_mask= (const float*)d_in[5];
  const float* aW1 = (const float*)d_in[6];
  const float* ab1 = (const float*)d_in[7];
  const float* aW2 = (const float*)d_in[8];
  const float* ab2 = (const float*)d_in[9];
  const float* aW3 = (const float*)d_in[10];
  const float* ab3 = (const float*)d_in[11];
  const float* vW1 = (const float*)d_in[12];
  const float* vb1 = (const float*)d_in[13];
  const float* vW2 = (const float*)d_in[14];
  const float* vb2 = (const float*)d_in[15];
  const float* cW1 = (const float*)d_in[16];
  const float* cb1 = (const float*)d_in[17];
  const float* cW2 = (const float*)d_in[18];
  const float* cb2 = (const float*)d_in[19];

  // workspace layout (floats): HPa | HPv | HPc | HLa | HLc | f16 B-fragments
  float* ws  = (float*)d_ws;
  float* HPa = ws;
  float* HPv = HPa + (size_t)BS*NPR*HD;
  float* HPc = HPv + (size_t)BS*NPR*HD;
  float* HLa = HPc + (size_t)BS*NPR*HD;
  float* HLc = HLa + (size_t)BS*NL*HD;
  _Float16* wsB = (_Float16*)(HLc + (size_t)BS*NL*HD);

  hipMemsetAsync(d_out, 0, (size_t)out_size*sizeof(float), stream);

  egnn_row_precompute<<<BS*NPR + BS*NL, 128, 0, stream>>>(
      h_ligand, h_protein, aW1, ab1, vW1, vb1, cW1, cb1,
      HPa, HPv, HPc, HLa, HLc);

  egnn_swizzle_b<<<8, 256, 0, stream>>>(aW2, vW2, wsB);

  egnn_cross_main<<<BS*(NL/16)*4, 256, 0, stream>>>(
      x_ligand, x_protein, ligand_mask, protein_mask,
      aW1, ab2, aW3, ab3, vW1, vb2, cW1, cW2, cb2,
      HPa, HPv, HPc, HLa, HLc, wsB, (float*)d_out);
}
